// MultiheadAttention_34248069218899
// MI455X (gfx1250) — matrix operand order repaired and checked by reading, not compiled, not run on hardware
//
#include <hip/hip_runtime.h>

// ---------------------------------------------------------------------------
// MHA for MI455X (gfx1250, wave32). bf16 WMMA with f32 accumulation.
// Pipeline: cvt(f32->bf16) -> QKV GEMMs -> flash-attention -> out-proj GEMM.
// Attention stages K via global_load_async_to_lds_b128 and V (transposed)
// into LDS, shared by all 8 waves of the block; all WMMA fragments are
// conflict-free ds_load_b128 reads (72-element padded row stride).
// ---------------------------------------------------------------------------

#define DIMSZ    1024
#define HEADS    16
#define HDIM     64
#define BATCH    4
#define SEQ      2048
#define MROWS    (BATCH * SEQ)   // 8192
#define KPAD     72              // padded LDS row stride (elements); banks (36n)%64 distinct

typedef __attribute__((ext_vector_type(16))) __bf16          v16bf;
typedef __attribute__((ext_vector_type(8)))  float           v8f;
typedef __attribute__((ext_vector_type(8)))  unsigned short  ushort8;
typedef __attribute__((ext_vector_type(16))) unsigned short  ushort16;

__device__ __forceinline__ unsigned short f32_to_bf16_bits(float f) {
    unsigned int u = __builtin_bit_cast(unsigned int, f);
    u += 0x7FFFu + ((u >> 16) & 1u);     // round-to-nearest-even
    return (unsigned short)(u >> 16);
}

__device__ __forceinline__ v8f v8f_zero() {
    v8f z = {0.f, 0.f, 0.f, 0.f, 0.f, 0.f, 0.f, 0.f};
    return z;
}

// A-fragment (16x32 bf16, M x K), row-major source, ld in elements.
// Layout (ISA 7.12.2): lane m = lane&15; half = lane>>4.
//   elements 0..7  = K in [half*8,      half*8+8)
//   elements 8..15 = K in [16 + half*8, 16 + half*8 + 8)
__device__ __forceinline__ v16bf load_a_frag(const unsigned short* base, int ld,
                                             int mBase, int k0, int lane) {
    const int half = (lane >> 4) & 1;
    const int m    = lane & 15;
    const unsigned short* p = base + (size_t)(mBase + m) * ld + k0 + half * 8;
    ushort8 lo = *(const ushort8*)(p);
    ushort8 hi = *(const ushort8*)(p + 16);
    ushort16 r;
#pragma unroll
    for (int t = 0; t < 8; ++t) { r[t] = lo[t]; r[8 + t] = hi[t]; }
    return __builtin_bit_cast(v16bf, r);
}

// B-fragment (32x16 bf16, K x N) where B[k][n] = src[(nBase+n)*ld + k0 + k].
// Layout: lane n = lane&15; half = lane>>4; elements 0..7 = K [half*8, +8), 8..15 = K [16+half*8, +8).
__device__ __forceinline__ v16bf load_b_frag_wt(const unsigned short* base, int ld,
                                                int nBase, int k0, int lane) {
    const int n  = lane & 15;
    const int ks = ((lane >> 4) & 1) * 8;
    const unsigned short* p = base + (size_t)(nBase + n) * ld + k0 + ks;
    ushort8 lo = *(const ushort8*)(p);
    ushort8 hi = *(const ushort8*)(p + 16);
    ushort16 r;
#pragma unroll
    for (int t = 0; t < 8; ++t) { r[t] = lo[t]; r[8 + t] = hi[t]; }
    return __builtin_bit_cast(v16bf, r);
}

// ---------------------------------------------------------------------------
__global__ void cvt_f32_bf16(const float* __restrict__ src,
                             unsigned short* __restrict__ dst, int n) {
    int i = blockIdx.x * blockDim.x + threadIdx.x;
    if (i < n) dst[i] = f32_to_bf16_bits(src[i]);
}

// ---------------------------------------------------------------------------
// GEMM: out = A @ W^T (+ bias). A: MxK bf16 row-major. W: NxK bf16 row-major.
// Each wave computes a 32x64 output tile. 8 waves per block.
// outf != nullptr -> f32 output with bias; else bf16 output to outb.
__global__ __launch_bounds__(256) void gemm_xwt(
        const unsigned short* __restrict__ A,
        const unsigned short* __restrict__ W,
        unsigned short* __restrict__ outb,
        float* __restrict__ outf,
        const float* __restrict__ bias,
        int M, int Ndim, int Kdim) {
    const int lane   = threadIdx.x & 31;
    const int wave   = threadIdx.x >> 5;
    const int tilesN = Ndim / 64;
    const int tile   = blockIdx.x * 8 + wave;
    if (tile >= (M / 32) * tilesN) return;
    const int mBase = (tile / tilesN) * 32;
    const int nBase = (tile % tilesN) * 64;

    v8f acc[2][4];
#pragma unroll
    for (int i = 0; i < 2; ++i)
#pragma unroll
        for (int j = 0; j < 4; ++j) acc[i][j] = v8f_zero();

    for (int k0 = 0; k0 < Kdim; k0 += 32) {
        v16bf a0 = load_a_frag(A, Kdim, mBase,      k0, lane);
        v16bf a1 = load_a_frag(A, Kdim, mBase + 16, k0, lane);
#pragma unroll
        for (int j = 0; j < 4; ++j) {
            v16bf b = load_b_frag_wt(W, Kdim, nBase + j * 16, k0, lane);
            acc[0][j] = __builtin_amdgcn_wmma_f32_16x16x32_bf16(
                false, a0, false, b, (short)0, acc[0][j], false, false);
            acc[1][j] = __builtin_amdgcn_wmma_f32_16x16x32_bf16(
                false, a1, false, b, (short)0, acc[1][j], false, false);
        }
    }

    // C layout: VGPR r, lane -> (M = r + (lane>>4)*8, N = lane&15)
    const int half = (lane >> 4) & 1;
    const int n    = lane & 15;
#pragma unroll
    for (int i = 0; i < 2; ++i)
#pragma unroll
        for (int j = 0; j < 4; ++j)
#pragma unroll
            for (int r = 0; r < 8; ++r) {
                const int row = mBase + i * 16 + half * 8 + r;
                const int col = nBase + j * 16 + n;
                float v = acc[i][j][r];
                if (outf) outf[(size_t)row * Ndim + col] = v + (bias ? bias[col] : 0.f);
                else      outb[(size_t)row * Ndim + col] = f32_to_bf16_bits(v);
            }
}

// ---------------------------------------------------------------------------
// Flash attention, block-cooperative: 8 waves of the block share one (b,h)
// and 128 query rows; each key block of 64 is staged once in LDS and reused
// by all 8 waves. K staged with async-to-LDS; V staged transposed.
__global__ __launch_bounds__(256) void attn_kernel(
        const unsigned short* __restrict__ Q,
        const unsigned short* __restrict__ K,
        const unsigned short* __restrict__ V,
        unsigned short* __restrict__ O) {
    __shared__ unsigned short Kt[64 * KPAD];        // [key][hd]   9216 B
    __shared__ unsigned short Vt[64 * KPAD];        // [hd][key]   9216 B (transposed)
    __shared__ unsigned short Pt[8][16 * KPAD];     // per-wave P 18432 B

    const int tid  = threadIdx.x;
    const int lane = tid & 31;
    const int wave = tid >> 5;

    const int qBlk = blockIdx.x & (SEQ / 128 - 1);          // 16 blocks per (b,h)
    const int h    = (blockIdx.x >> 4) & (HEADS - 1);
    const int b    = blockIdx.x >> 8;
    const int qBase = qBlk * 128 + wave * 16;
    const float scale = 0.125f;                              // HDIM^-0.5

    const size_t headOff = (size_t)b * SEQ * DIMSZ + (size_t)h * HDIM;
    const unsigned short* Qh = Q + headOff;
    const unsigned short* Kh = K + headOff;
    const unsigned short* Vh = V + headOff;

    // Q fragments: 16 x 64 over head_dim, two K-steps of 32
    v16bf q0 = load_a_frag(Qh, DIMSZ, qBase, 0,  lane);
    v16bf q1 = load_a_frag(Qh, DIMSZ, qBase, 32, lane);

    float m_i[8], l_i[8];
#pragma unroll
    for (int r = 0; r < 8; ++r) { m_i[r] = -3.0e38f; l_i[r] = 0.f; }
    v8f oacc[4];
#pragma unroll
    for (int t = 0; t < 4; ++t) oacc[t] = v8f_zero();

    const int half = (lane >> 4) & 1;
    const int nl   = lane & 15;
    unsigned short* myP = Pt[wave];

    for (int kb = 0; kb < SEQ; kb += 64) {
        __syncthreads();   // all waves finished reading previous K/V tiles

        // ---- stage K tile [64 keys x 64 hd] via async-to-LDS (16B per op) ----
        // 512 chunks of 8 elements; each thread issues 2 async b128 copies.
#pragma unroll
        for (int it = 0; it < 2; ++it) {
            const int c   = tid + it * 256;
            const int row = c >> 3;
            const int col = (c & 7) * 8;
            unsigned long long gaddr =
                (unsigned long long)(const void*)(Kh + (size_t)(kb + row) * DIMSZ + col);
            unsigned int ldsOff =
                (unsigned int)(unsigned long long)(const void*)(&Kt[row * KPAD + col]);
            asm volatile("global_load_async_to_lds_b128 %0, %1, off"
                         :: "v"(ldsOff), "v"(gaddr) : "memory");
        }

        // ---- stage V tile transposed: Vt[hd][key] (coalesced global b128) ----
#pragma unroll
        for (int it = 0; it < 2; ++it) {
            const int c   = tid + it * 256;
            const int row = c >> 3;                 // key
            const int col = (c & 7) * 8;            // hd base
            ushort8 vv = *(const ushort8*)(Vh + (size_t)(kb + row) * DIMSZ + col);
#pragma unroll
            for (int e = 0; e < 8; ++e)
                Vt[(col + e) * KPAD + row] = vv[e];
        }

        asm volatile("s_wait_asynccnt 0" ::: "memory");
        __syncthreads();   // staged tiles visible to all waves

        // ---- S = scale * Q @ K^T for 4 key tiles of 16 (LDS b128 frags) ----
        v8f s[4];
#pragma unroll
        for (int j = 0; j < 4; ++j) {
            v16bf bk0 = load_b_frag_wt(Kt, KPAD, j * 16, 0,  lane);
            v16bf bk1 = load_b_frag_wt(Kt, KPAD, j * 16, 32, lane);
            v8f z = v8f_zero();
            z = __builtin_amdgcn_wmma_f32_16x16x32_bf16(false, q0, false, bk0,
                                                        (short)0, z, false, false);
            z = __builtin_amdgcn_wmma_f32_16x16x32_bf16(false, q1, false, bk1,
                                                        (short)0, z, false, false);
            s[j] = z * scale;
        }

        // ---- online softmax per row r (row M = r + half*8; N over 16 lanes) ----
        float alpha[8];
#pragma unroll
        for (int r = 0; r < 8; ++r) {
            float blk = fmaxf(fmaxf(s[0][r], s[1][r]), fmaxf(s[2][r], s[3][r]));
#pragma unroll
            for (int msk = 1; msk < 16; msk <<= 1)
                blk = fmaxf(blk, __shfl_xor(blk, msk, 32));
            const float mnew = fmaxf(m_i[r], blk);
            alpha[r] = __expf(m_i[r] - mnew);
            float rs = 0.f;
#pragma unroll
            for (int j = 0; j < 4; ++j) {
                float pv = __expf(s[j][r] - mnew);
                s[j][r] = pv;
                rs += pv;
            }
#pragma unroll
            for (int msk = 1; msk < 16; msk <<= 1)
                rs += __shfl_xor(rs, msk, 32);
            l_i[r] = l_i[r] * alpha[r] + rs;
            m_i[r] = mnew;
        }
#pragma unroll
        for (int t = 0; t < 4; ++t)
#pragma unroll
            for (int r = 0; r < 8; ++r) oacc[t][r] *= alpha[r];

        // ---- stage P (bf16) in per-wave LDS, reload as A-fragments ----
#pragma unroll
        for (int j = 0; j < 4; ++j)
#pragma unroll
            for (int r = 0; r < 8; ++r)
                myP[(half * 8 + r) * KPAD + j * 16 + nl] = f32_to_bf16_bits(s[j][r]);

        v16bf p0 = load_a_frag(myP, KPAD, 0, 0,  lane);
        v16bf p1 = load_a_frag(myP, KPAD, 0, 32, lane);

        // ---- O += P @ V : B-frags are contiguous rows of transposed Vt ----
#pragma unroll
        for (int t = 0; t < 4; ++t) {
            v16bf bv0 = load_b_frag_wt(Vt, KPAD, t * 16, 0,  lane);
            v16bf bv1 = load_b_frag_wt(Vt, KPAD, t * 16, 32, lane);
            oacc[t] = __builtin_amdgcn_wmma_f32_16x16x32_bf16(
                false, p0, false, bv0, (short)0, oacc[t], false, false);
            oacc[t] = __builtin_amdgcn_wmma_f32_16x16x32_bf16(
                false, p1, false, bv1, (short)0, oacc[t], false, false);
        }
    }

    // ---- normalize and store O (bf16, [b,n,h,hd] row-major) ----
    unsigned short* Oh = O + headOff;
#pragma unroll
    for (int t = 0; t < 4; ++t)
#pragma unroll
        for (int r = 0; r < 8; ++r) {
            const int row = qBase + half * 8 + r;
            float v = oacc[t][r] / l_i[r];
            Oh[(size_t)row * DIMSZ + t * 16 + nl] = f32_to_bf16_bits(v);
        }
}

// ---------------------------------------------------------------------------
extern "C" void kernel_launch(void* const* d_in, const int* in_sizes, int n_in,
                              void* d_out, int out_size, void* d_ws, size_t ws_size,
                              hipStream_t stream) {
    const float* x  = (const float*)d_in[0];
    const float* Wq = (const float*)d_in[1];
    const float* Wk = (const float*)d_in[2];
    const float* Wv = (const float*)d_in[3];
    const float* Wo = (const float*)d_in[4];
    const float* bo = (const float*)d_in[5];

    char* ws = (char*)d_ws;
    size_t off = 0;
    auto carve = [&](size_t bytes) -> unsigned short* {
        unsigned short* p = (unsigned short*)(ws + off);
        off += (bytes + 255) & ~(size_t)255;
        return p;
    };
    const size_t xElems = (size_t)MROWS * DIMSZ;     // 8M
    const size_t wElems = (size_t)DIMSZ * DIMSZ;     // 1M
    unsigned short* xb  = carve(xElems * 2);
    unsigned short* wqb = carve(wElems * 2);
    unsigned short* wkb = carve(wElems * 2);
    unsigned short* wvb = carve(wElems * 2);
    unsigned short* wob = carve(wElems * 2);
    unsigned short* Qb  = carve(xElems * 2);
    unsigned short* Kb  = carve(xElems * 2);
    unsigned short* Vb  = carve(xElems * 2);
    unsigned short* Ob  = carve(xElems * 2);

    // 1) fp32 -> bf16 conversions
    cvt_f32_bf16<<<(int)((xElems + 255) / 256), 256, 0, stream>>>(x,  xb,  (int)xElems);
    cvt_f32_bf16<<<(int)((wElems + 255) / 256), 256, 0, stream>>>(Wq, wqb, (int)wElems);
    cvt_f32_bf16<<<(int)((wElems + 255) / 256), 256, 0, stream>>>(Wk, wkb, (int)wElems);
    cvt_f32_bf16<<<(int)((wElems + 255) / 256), 256, 0, stream>>>(Wv, wvb, (int)wElems);
    cvt_f32_bf16<<<(int)((wElems + 255) / 256), 256, 0, stream>>>(Wo, wob, (int)wElems);

    // 2) Q/K/V projections: (8192x1024) @ (1024x1024)^T, bf16 out
    const int gemmBlocks = (MROWS / 32) * (DIMSZ / 64) / 8;   // 512
    gemm_xwt<<<gemmBlocks, 256, 0, stream>>>(xb, wqb, Qb, nullptr, nullptr,
                                             MROWS, DIMSZ, DIMSZ);
    gemm_xwt<<<gemmBlocks, 256, 0, stream>>>(xb, wkb, Kb, nullptr, nullptr,
                                             MROWS, DIMSZ, DIMSZ);
    gemm_xwt<<<gemmBlocks, 256, 0, stream>>>(xb, wvb, Vb, nullptr, nullptr,
                                             MROWS, DIMSZ, DIMSZ);

    // 3) attention: 8 waves per block share one (b,h), 128 query rows/block
    const int attnBlocks = BATCH * HEADS * (SEQ / 128);       // 1024
    attn_kernel<<<attnBlocks, 256, 0, stream>>>(Qb, Kb, Vb, Ob);

    // 4) output projection with bias, f32 out
    gemm_xwt<<<gemmBlocks, 256, 0, stream>>>(Ob, wob, nullptr, (float*)d_out, bo,
                                             MROWS, DIMSZ, DIMSZ);
}
